// MultiHeadSelfAttention_22058952032294
// MI455X (gfx1250) — matrix operand order repaired and checked by reading, not compiled, not run on hardware
//
#include <hip/hip_runtime.h>

typedef unsigned short u16;
typedef unsigned int   u32;
typedef __attribute__((ext_vector_type(4)))  u16    v4us;
typedef __attribute__((ext_vector_type(8)))  u16    v8us;
typedef __attribute__((ext_vector_type(8)))  __bf16 v8bf;
typedef __attribute__((ext_vector_type(16))) __bf16 v16bf;
typedef __attribute__((ext_vector_type(8)))  float  v8f;

namespace {
constexpr int kB = 2, kS = 2048, kD = 1024, kH = 16, kDK = 64;
constexpr int kBS = kB * kS;
constexpr int kWST = 40;            // LDS weight-tile row stride (u16): 80B, 16B-aligned, bank-spread
}

union AF { v16bf v; v8bf h[2]; };

__device__ __forceinline__ u16 f2bf(float f) {
  u32 u = __builtin_bit_cast(u32, f);
  u = u + 0x7FFFu + ((u >> 16) & 1u);   // round-to-nearest-even
  return (u16)(u >> 16);
}

__device__ __forceinline__ v8bf ld8(const u16* p) { return *(const v8bf*)p; }

__device__ __forceinline__ v8f vzero() {
  v8f z;
#pragma unroll
  for (int i = 0; i < 8; ++i) z[i] = 0.f;
  return z;
}

__device__ __forceinline__ v8f bwmma(const AF& a, const AF& b, v8f c) {
  return __builtin_amdgcn_wmma_f32_16x16x32_bf16(false, a.v, false, b.v,
                                                 (short)0, c, false, false);
}

// Async copy of one 16B chunk from global to LDS (ASYNCcnt-tracked, GVS mode).
__device__ __forceinline__ void async_g2l_b128(u32 lds_byte_addr, u32 voff_bytes,
                                               const u16* sbase) {
  asm volatile("global_load_async_to_lds_b128 %0, %1, %2"
               :: "v"(lds_byte_addr), "v"(voff_bytes), "s"(sbase)
               : "memory");
}

__device__ __forceinline__ void wait_async0() {
  asm volatile("s_wait_asynccnt 0x0" ::: "memory");
}

// ---------------------------------------------------------------- stage 1
__global__ void __launch_bounds__(256) cvt_kernel(const float* __restrict__ in,
                                                  u16* __restrict__ out, int n4) {
  int i = blockIdx.x * 256 + threadIdx.x;
  if (i >= n4) return;
  float4 f = ((const float4*)in)[i];
  v4us o;
  o[0] = f2bf(f.x); o[1] = f2bf(f.y); o[2] = f2bf(f.z); o[3] = f2bf(f.w);
  ((v4us*)out)[i] = o;
}

// ---------------------------------------------------------------- stage 2
// Fused QKV projection (bf16 WMMA, fp32 acc) + RoPE epilogue.
// Weight tiles (64 rows x 32 k, shared by all 8 waves) are double-buffered in
// LDS via global_load_async_to_lds_b128.
// qo/ko layout: [b][h][s][dk].  vo layout (transposed): [b][h][dk][s].
__global__ void __launch_bounds__(256) qkv_rope_kernel(
    const u16* __restrict__ xb,
    const u16* __restrict__ wq, const u16* __restrict__ wk, const u16* __restrict__ wv,
    const float* __restrict__ sin_tab, const float* __restrict__ cos_tab,
    const int* __restrict__ tpos,
    u16* __restrict__ qo, u16* __restrict__ ko, u16* __restrict__ vo) {
  __shared__ __align__(16) u16 Wlds[2][3][64 * kWST];   // 30 KB
  const int t = threadIdx.x;
  const int lane = t & 31;
  const int wave = t >> 5;
  const int m = lane & 15;
  const int g = lane >> 4;
  const int rowBase = (blockIdx.x * 8 + wave) * 16;   // token-row tile (over B*S)
  const int colBase = blockIdx.y * 64;                // output-dim tile

  // staging role: 256 threads x 3 matrices x 16B = 64x32 bf16 tile per matrix
  const int srow = t >> 2;                 // 0..63
  const int scol = (t & 3) * 8;            // 0,8,16,24
  const u32 voff = (u32)(srow * kD + scol) * 2u;
  const u32 ldsoff = (u32)(srow * kWST + scol) * 2u;
  const u16* wsrc[3] = {wq, wk, wv};

  auto stage = [&](int buf, int k0) {
#pragma unroll
    for (int j = 0; j < 3; ++j) {
      const u16* gb = wsrc[j] + (size_t)colBase * kD + k0;
      const u32 dst = (u32)(uintptr_t)(&Wlds[buf][j][0]) + ldsoff;
      async_g2l_b128(dst, voff, gb);
    }
  };

  v8f aq[4], ak[4], av[4];
#pragma unroll
  for (int i = 0; i < 4; ++i) { aq[i] = vzero(); ak[i] = vzero(); av[i] = vzero(); }

  const u16* xrow = xb + (size_t)(rowBase + m) * kD;
  stage(0, 0);
  for (int k0 = 0; k0 < kD; k0 += 32) {
    const int buf = (k0 >> 5) & 1;
    wait_async0();            // own async transfers into `buf` have landed
    __syncthreads();          // all waves' transfers into `buf` have landed
    if (k0 + 32 < kD) stage(buf ^ 1, k0 + 32);

    AF a;                                   // A frag: K = e + 16g
    a.h[0] = ld8(xrow + k0 + g * 16);
    a.h[1] = ld8(xrow + k0 + g * 16 + 8);
    __builtin_prefetch(xrow + k0 + 64, 0, 1);
#pragma unroll
    for (int nf = 0; nf < 4; ++nf) {
      const int nl = nf * 16 + m;           // local B row; K = e + 16g
      const u16* pq = &Wlds[buf][0][nl * kWST + g * 16];
      const u16* pk = &Wlds[buf][1][nl * kWST + g * 16];
      const u16* pv = &Wlds[buf][2][nl * kWST + g * 16];
      AF bq, bk, bv;
      bq.h[0] = ld8(pq); bq.h[1] = ld8(pq + 8);
      bk.h[0] = ld8(pk); bk.h[1] = ld8(pk + 8);
      bv.h[0] = ld8(pv); bv.h[1] = ld8(pv + 8);
      aq[nf] = bwmma(a, bq, aq[nf]);
      ak[nf] = bwmma(a, bk, ak[nf]);
      av[nf] = bwmma(a, bv, av[nf]);
    }
    __syncthreads();          // retire reads of `buf` before it is restaged
  }

  // --- V epilogue: store transposed, 8 contiguous tokens per lane (b128 store)
#pragma unroll
  for (int nf = 0; nf < 4; ++nf) {
    const int od = colBase + nf * 16 + m;
    const int hh = od >> 6, dk = od & 63;
    const int tok0 = rowBase + 8 * g;
    const int b = tok0 >> 11;
    const int s0 = tok0 & (kS - 1);
    v8us vv;
#pragma unroll
    for (int r = 0; r < 8; ++r) vv[r] = f2bf(av[nf][r]);
    *(v8us*)(vo + ((size_t)(b * kH + hh) * kDK + dk) * kS + s0) = vv;
  }

  // --- Q/K epilogue with RoPE: even/odd dk pair lives in adjacent lanes
#pragma unroll
  for (int nf = 0; nf < 4; ++nf) {
    const int od = colBase + nf * 16 + m;
    const int hh = od >> 6, dk = od & 63;
    const int pidx = dk >> 1;
    const bool even = (dk & 1) == 0;
#pragma unroll
    for (int r = 0; r < 8; ++r) {
      const int tok = rowBase + 8 * g + r;
      const int b = tok >> 11;
      const int s = tok & (kS - 1);
      const int pos = tpos[b * kS + s];
      const float sv = sin_tab[pos * (kDK / 2) + pidx];
      const float cv = cos_tab[pos * (kDK / 2) + pidx];
      float qs = aq[nf][r];
      float qp = __shfl_xor(qs, 1, 32);
      float ks = ak[nf][r];
      float kp = __shfl_xor(ks, 1, 32);
      float qr = even ? (cv * qs - sv * qp) : (cv * qs + sv * qp);
      float kr = even ? (cv * ks - sv * kp) : (cv * ks + sv * kp);
      const size_t base = ((size_t)(b * kH + hh) * kS + s) * kDK + dk;
      qo[base] = f2bf(qr);
      ko[base] = f2bf(kr);
    }
  }
}

// ---------------------------------------------------------------- stage 3
// Causal flash attention: one 16-query tile per wave, 32-key chunks.
__global__ void __launch_bounds__(256) flash_attn_kernel(
    const u16* __restrict__ qb, const u16* __restrict__ kb,
    const u16* __restrict__ vtb, u16* __restrict__ ab) {
  __shared__ __align__(16) u16 Plds[8][16 * 32];
  const int lane = threadIdx.x & 31;
  const int wave = threadIdx.x >> 5;
  const int m = lane & 15;
  const int g = lane >> 4;
  const int wid = blockIdx.x * 8 + wave;
  const int qt = wid & (kS / 16 - 1);
  const int bh = wid >> 7;            // S/16 = 128 tiles per (b,h)
  const int qBase = qt * 16;

  AF qa0, qa1;                        // Q A-frags, K-dim = dk (0..63)
  const u16* qp = qb + ((size_t)bh * kS + qBase + m) * kDK;
  qa0.h[0] = ld8(qp + g * 16);       qa0.h[1] = ld8(qp + 8 + g * 16);
  qa1.h[0] = ld8(qp + 32 + g * 16);  qa1.h[1] = ld8(qp + 40 + g * 16);

  float mrun[8], lrun[8], corr[8];
  v8f o[4];
#pragma unroll
  for (int r = 0; r < 8; ++r) { mrun[r] = -1e30f; lrun[r] = 0.f; }
#pragma unroll
  for (int i = 0; i < 4; ++i) o[i] = vzero();

  u16* pl = &Plds[wave][0];
  const u16* kbase = kb + (size_t)bh * kS * kDK;
  const u16* vbase = vtb + (size_t)bh * kDK * kS;

  for (int c0 = 0; c0 <= qBase + 15; c0 += 32) {
    // ---- scores S = Q K^T (two 16-key frags, K-dim 64 split 2x32)
    v8f s0 = vzero(), s1 = vzero();
    {
      AF bk;
      const u16* kp0 = kbase + (size_t)(c0 + m) * kDK + g * 16;
      const u16* kp1 = kbase + (size_t)(c0 + 16 + m) * kDK + g * 16;
      bk.h[0] = ld8(kp0);      bk.h[1] = ld8(kp0 + 8);   s0 = bwmma(qa0, bk, s0);
      bk.h[0] = ld8(kp0 + 32); bk.h[1] = ld8(kp0 + 40);  s0 = bwmma(qa1, bk, s0);
      bk.h[0] = ld8(kp1);      bk.h[1] = ld8(kp1 + 8);   s1 = bwmma(qa0, bk, s1);
      bk.h[0] = ld8(kp1 + 32); bk.h[1] = ld8(kp1 + 40);  s1 = bwmma(qa1, bk, s1);
    }
    // ---- online softmax (row reductions across the 16-lane group)
#pragma unroll
    for (int r = 0; r < 8; ++r) {
      const int qrow = qBase + 8 * g + r;
      float a0 = (c0 + m      > qrow) ? -1e30f : s0[r] * 0.125f;
      float a1 = (c0 + 16 + m > qrow) ? -1e30f : s1[r] * 0.125f;
      float t = fmaxf(a0, a1);
#pragma unroll
      for (int off = 1; off < 16; off <<= 1) t = fmaxf(t, __shfl_xor(t, off, 32));
      const float mn = fmaxf(mrun[r], t);
      const float c = __expf(mrun[r] - mn);
      const float p0 = __expf(a0 - mn);
      const float p1 = __expf(a1 - mn);
      float sum = p0 + p1;
#pragma unroll
      for (int off = 1; off < 16; off <<= 1) sum += __shfl_xor(sum, off, 32);
      lrun[r] = lrun[r] * c + sum;
      mrun[r] = mn;
      corr[r] = c;
      const int row = 8 * g + r;        // transpose P through LDS (D- -> A-layout)
      pl[row * 32 + m]      = f2bf(p0);
      pl[row * 32 + 16 + m] = f2bf(p1);
    }
#pragma unroll
    for (int nf = 0; nf < 4; ++nf)
#pragma unroll
      for (int r = 0; r < 8; ++r) o[nf][r] *= corr[r];

    AF pa;                              // P A-frag (K-dim = 32 keys)
    pa.h[0] = ld8(pl + m * 32 + g * 16);
    pa.h[1] = ld8(pl + m * 32 + g * 16 + 8);
#pragma unroll
    for (int nf = 0; nf < 4; ++nf) {    // V B-frag: contiguous keys via Vt layout
      const u16* vp = vbase + (size_t)(nf * 16 + m) * kS + c0 + g * 16;
      AF bv; bv.h[0] = ld8(vp); bv.h[1] = ld8(vp + 8);
      o[nf] = bwmma(pa, bv, o[nf]);
    }
  }

  const int b = bh >> 4;
  const int hh = bh & 15;
#pragma unroll
  for (int nf = 0; nf < 4; ++nf)
#pragma unroll
    for (int r = 0; r < 8; ++r) {
      const int s = qBase + 8 * g + r;
      const float val = o[nf][r] / lrun[r];
      ab[((size_t)b * kS + s) * kD + hh * 64 + nf * 16 + m] = f2bf(val);
    }
}

// ---------------------------------------------------------------- stage 4
__global__ void __launch_bounds__(256) oproj_kernel(
    const u16* __restrict__ ab, const u16* __restrict__ wo,
    float* __restrict__ out) {
  __shared__ __align__(16) u16 Olds[2][64 * kWST];      // 10 KB
  const int t = threadIdx.x;
  const int lane = t & 31;
  const int wave = t >> 5;
  const int m = lane & 15;
  const int g = lane >> 4;
  const int rowBase = (blockIdx.x * 8 + wave) * 16;
  const int colBase = blockIdx.y * 64;

  const int srow = t >> 2;
  const int scol = (t & 3) * 8;
  const u32 voff = (u32)(srow * kD + scol) * 2u;
  const u32 ldsoff = (u32)(srow * kWST + scol) * 2u;

  auto stage = [&](int buf, int k0) {
    const u16* gb = wo + (size_t)colBase * kD + k0;
    const u32 dst = (u32)(uintptr_t)(&Olds[buf][0]) + ldsoff;
    async_g2l_b128(dst, voff, gb);
  };

  v8f acc[4];
#pragma unroll
  for (int i = 0; i < 4; ++i) acc[i] = vzero();

  const u16* arow = ab + (size_t)(rowBase + m) * kD;
  stage(0, 0);
  for (int k0 = 0; k0 < kD; k0 += 32) {
    const int buf = (k0 >> 5) & 1;
    wait_async0();
    __syncthreads();
    if (k0 + 32 < kD) stage(buf ^ 1, k0 + 32);

    AF a;
    a.h[0] = ld8(arow + k0 + g * 16);
    a.h[1] = ld8(arow + k0 + g * 16 + 8);
    __builtin_prefetch(arow + k0 + 64, 0, 1);
#pragma unroll
    for (int nf = 0; nf < 4; ++nf) {
      const int nl = nf * 16 + m;
      const u16* pw = &Olds[buf][nl * kWST + g * 16];
      AF bw; bw.h[0] = ld8(pw); bw.h[1] = ld8(pw + 8);
      acc[nf] = bwmma(a, bw, acc[nf]);
    }
    __syncthreads();
  }
#pragma unroll
  for (int nf = 0; nf < 4; ++nf)
#pragma unroll
    for (int r = 0; r < 8; ++r)
      out[(size_t)(rowBase + 8 * g + r) * kD + colBase + nf * 16 + m] = acc[nf][r];
}

// ---------------------------------------------------------------- launcher
extern "C" void kernel_launch(void* const* d_in, const int* in_sizes, int n_in,
                              void* d_out, int out_size, void* d_ws, size_t ws_size,
                              hipStream_t stream) {
  (void)in_sizes; (void)n_in; (void)out_size; (void)ws_size;
  const float* x    = (const float*)d_in[0];
  const float* w_q  = (const float*)d_in[1];
  const float* w_k  = (const float*)d_in[2];
  const float* w_v  = (const float*)d_in[3];
  const float* w_o  = (const float*)d_in[4];
  const float* sint = (const float*)d_in[5];
  const float* cost = (const float*)d_in[6];
  const int*   tpos = (const int*)d_in[7];
  float* out = (float*)d_out;

  char* ws = (char*)d_ws;                    // 48 MB total
  u16* xb  = (u16*)(ws);                     // 8 MB  x bf16
  u16* wqb = (u16*)(ws + (8ull  << 20));     // 2 MB each
  u16* wkb = (u16*)(ws + (10ull << 20));
  u16* wvb = (u16*)(ws + (12ull << 20));
  u16* wob = (u16*)(ws + (14ull << 20));
  u16* qb  = (u16*)(ws + (16ull << 20));     // 8 MB  Q roped [b][h][s][dk]
  u16* kb  = (u16*)(ws + (24ull << 20));     // 8 MB  K roped [b][h][s][dk]
  u16* vtb = (u16*)(ws + (32ull << 20));     // 8 MB  V^T     [b][h][dk][s]
  u16* ab  = (u16*)(ws + (40ull << 20));     // 8 MB  attn out [b][s][d]

  cvt_kernel<<<(kBS * kD / 4 + 255) / 256, 256, 0, stream>>>(x,   xb,  kBS * kD / 4);
  cvt_kernel<<<(kD * kD / 4 + 255) / 256, 256, 0, stream>>>(w_q, wqb, kD * kD / 4);
  cvt_kernel<<<(kD * kD / 4 + 255) / 256, 256, 0, stream>>>(w_k, wkb, kD * kD / 4);
  cvt_kernel<<<(kD * kD / 4 + 255) / 256, 256, 0, stream>>>(w_v, wvb, kD * kD / 4);
  cvt_kernel<<<(kD * kD / 4 + 255) / 256, 256, 0, stream>>>(w_o, wob, kD * kD / 4);

  dim3 gGemm(kBS / 128, kD / 64);            // (32, 16), 8 waves/block
  qkv_rope_kernel<<<gGemm, 256, 0, stream>>>(xb, wqb, wkb, wvb, sint, cost, tpos,
                                             qb, kb, vtb);
  flash_attn_kernel<<<(kB * kH * (kS / 16)) / 8, 256, 0, stream>>>(qb, kb, vtb, ab);
  oproj_kernel<<<gGemm, 256, 0, stream>>>(ab, wob, out);
}
